// Attention_26414048870779
// MI455X (gfx1250) — matrix operand order repaired and checked by reading, not compiled, not run on hardware
//
#include <hip/hip_runtime.h>
#include <hip/hip_bf16.h>

typedef __attribute__((ext_vector_type(16))) __bf16 v16bf;
typedef __attribute__((ext_vector_type(8)))  __bf16 v8bf;
typedef __attribute__((ext_vector_type(4)))  __bf16 v4bf;
typedef __attribute__((ext_vector_type(2)))  __bf16 v2bf;
typedef __attribute__((ext_vector_type(8)))  float  v8f;
typedef __attribute__((ext_vector_type(4)))  float  v4f;
typedef __attribute__((ext_vector_type(4)))  int    v4i;

#define EMBED 2048
#define DHEAD 256
#define BATCH 16
#define SEQ   2048

#ifndef __has_builtin
#define __has_builtin(x) 0
#endif

#if __has_builtin(__builtin_amdgcn_global_load_async_to_lds_b128)
#define HAVE_ASYNC_COPY 1
#else
#define HAVE_ASYNC_COPY 0
#endif

#if __has_builtin(__builtin_amdgcn_s_wait_asynccnt)
#define WAIT_ASYNC() __builtin_amdgcn_s_wait_asynccnt(0)
#else
#define WAIT_ASYNC() asm volatile("s_wait_asynccnt 0x0" ::: "memory")
#endif

static __device__ inline v16bf joinv(v8bf lo, v8bf hi) {
  return __builtin_shufflevector(lo, hi, 0,1,2,3,4,5,6,7,8,9,10,11,12,13,14,15);
}

static __device__ inline v8f zero8() {
  v8f z;
#pragma unroll
  for (int i = 0; i < 8; ++i) z[i] = 0.0f;
  return z;
}

static __device__ inline float lanemax16(float v, int lane) {
#pragma unroll
  for (int m = 1; m < 16; m <<= 1) {
    int o = __builtin_amdgcn_ds_bpermute((lane ^ m) << 2, __builtin_bit_cast(int, v));
    v = fmaxf(v, __builtin_bit_cast(float, o));
  }
  return v;
}

static __device__ inline float lanesum16(float v, int lane) {
#pragma unroll
  for (int m = 1; m < 16; m <<= 1) {
    int o = __builtin_amdgcn_ds_bpermute((lane ^ m) << 2, __builtin_bit_cast(int, v));
    v += __builtin_bit_cast(float, o);
  }
  return v;
}

// Issue copy of one 32x256 bf16 K tile into LDS (stride 264 halves).
// Async (ASYNCcnt-tracked, VGPR-bypassing) when the builtin exists.
static __device__ inline void issue_k_copy(const __bf16* __restrict__ Kp,
                                           __bf16* dst, int kb0, int tid) {
#pragma unroll
  for (int i = 0; i < 4; ++i) {
    int q = tid + i * 256;
    int row = q >> 5, dc = q & 31;
    const __bf16* g = Kp + (size_t)(kb0 + row) * DHEAD + dc * 8;
    __bf16* l = dst + row * 264 + dc * 8;
#if HAVE_ASYNC_COPY
    __builtin_amdgcn_global_load_async_to_lds_b128(
        (__attribute__((address_space(1))) v4i*)(void*)const_cast<__bf16*>(g),
        (__attribute__((address_space(3))) v4i*)(void*)l, 0, 0);
#else
    *reinterpret_cast<v8bf*>(l) = *reinterpret_cast<const v8bf*>(g);
#endif
  }
}

// V tile: global loads (pair of key rows per thread) -> registers
static __device__ inline void load_v_regs(const __bf16* __restrict__ Vp, int kb0,
                                          int tid, v8bf va[2], v8bf vb[2]) {
#pragma unroll
  for (int i = 0; i < 2; ++i) {
    int q = tid + i * 256;
    int kp = q >> 5, dc = q & 31;
    const __bf16* base = Vp + (size_t)(kb0 + 2 * kp) * DHEAD + dc * 8;
    va[i] = *reinterpret_cast<const v8bf*>(base);
    vb[i] = *reinterpret_cast<const v8bf*>(base + DHEAD);
  }
}

// registers -> transposed LDS [d][key] (stride 40), packed b32 stores
static __device__ inline void store_v_lds(__bf16* dst, int tid,
                                          const v8bf va[2], const v8bf vb[2]) {
#pragma unroll
  for (int i = 0; i < 2; ++i) {
    int q = tid + i * 256;
    int kp = q >> 5, dc = q & 31;
#pragma unroll
    for (int e = 0; e < 8; ++e) {
      v2bf t;
      t[0] = va[i][e];
      t[1] = vb[i][e];
      *reinterpret_cast<v2bf*>(&dst[(dc * 8 + e) * 40 + 2 * kp]) = t;
    }
  }
}

// ---------------------------------------------------------------------------
// Projection GEMM: Y[M,256](bf16) = X[M,2048](f32) @ W[2048,256](f32)
// ---------------------------------------------------------------------------
__global__ __launch_bounds__(256) void proj_kernel(
    const float* __restrict__ X, const float* __restrict__ W,
    __bf16* __restrict__ Y)
{
  __shared__ __bf16 As[128 * 40];  // [row][k], k padded 32->40
  __shared__ __bf16 Wt[64 * 40];   // transposed W tile: [n][k]

  const int tid  = threadIdx.x;
  const int lane = tid & 31;
  const int w    = tid >> 5;
  const int hi   = lane >> 4;
  const int ln   = lane & 15;
  const int m0   = blockIdx.x * 128;
  const int n0   = blockIdx.y * 64;

  v8f acc[4];
#pragma unroll
  for (int j = 0; j < 4; ++j) acc[j] = zero8();

  for (int kt = 0; kt < EMBED / 32; ++kt) {
    const int k0 = kt * 32;
    __syncthreads();
#pragma unroll
    for (int i = 0; i < 4; ++i) {
      int q = tid + i * 256;
      int row = q >> 3, c4 = q & 7;
      v4f vals = *reinterpret_cast<const v4f*>(
          &X[(size_t)(m0 + row) * EMBED + k0 + c4 * 4]);
      v4bf bv;
#pragma unroll
      for (int e = 0; e < 4; ++e) bv[e] = (__bf16)vals[e];
      *reinterpret_cast<v4bf*>(&As[row * 40 + c4 * 4]) = bv;
    }
#pragma unroll
    for (int i = 0; i < 2; ++i) {
      int q = tid + i * 256;
      int kk = q >> 4, c4 = q & 15;
      v4f vals = *reinterpret_cast<const v4f*>(
          &W[(size_t)(k0 + kk) * DHEAD + n0 + c4 * 4]);
#pragma unroll
      for (int e = 0; e < 4; ++e) Wt[(c4 * 4 + e) * 40 + kk] = (__bf16)vals[e];
    }
    __syncthreads();

    const int arow = 16 * w + ln;
    v8bf alo = *reinterpret_cast<const v8bf*>(&As[arow * 40 + hi * 8]);
    v8bf ahi = *reinterpret_cast<const v8bf*>(&As[arow * 40 + hi * 8 + 16]);
    v16bf afrag = joinv(alo, ahi);
#pragma unroll
    for (int j = 0; j < 4; ++j) {
      const int n = 16 * j + ln;
      v8bf blo = *reinterpret_cast<const v8bf*>(&Wt[n * 40 + hi * 8]);
      v8bf bhi = *reinterpret_cast<const v8bf*>(&Wt[n * 40 + hi * 8 + 16]);
      acc[j] = __builtin_amdgcn_wmma_f32_16x16x32_bf16(
          false, afrag, false, joinv(blo, bhi), (short)0, acc[j], false, false);
    }
  }

#pragma unroll
  for (int j = 0; j < 4; ++j) {
#pragma unroll
    for (int r = 0; r < 8; ++r) {
      int grow = m0 + 16 * w + r + 8 * hi;
      Y[(size_t)grow * DHEAD + n0 + 16 * j + ln] = (__bf16)acc[j][r];
    }
  }
}

// ---------------------------------------------------------------------------
// Flash attention, double-buffered 32-key tiles with async K copies.
// Block = 8 waves; wave owns 16 query rows x full D=256.
// ---------------------------------------------------------------------------
__global__ __launch_bounds__(256) void attn_kernel(
    const __bf16* __restrict__ Qb, const __bf16* __restrict__ Kb,
    const __bf16* __restrict__ Vb, float* __restrict__ Out)
{
  __shared__ __bf16 Ks[2][32 * 264];    // K tile [key][d], pad 256->264
  __shared__ __bf16 Vt[2][256 * 40];    // V tile transposed [d][key], pad 32->40
  __shared__ __bf16 Ps[8 * 16 * 40];    // per-wave P staging

  const int tid  = threadIdx.x;
  const int lane = tid & 31;
  const int w    = tid >> 5;
  const int hi   = lane >> 4;
  const int ln   = lane & 15;
  const int b    = blockIdx.y;
  const int m0w  = blockIdx.x * 128 + w * 16;

  const __bf16* Qp = Qb + (size_t)b * SEQ * DHEAD;
  const __bf16* Kp = Kb + (size_t)b * SEQ * DHEAD;
  const __bf16* Vp = Vb + (size_t)b * SEQ * DHEAD;

  // Q fragments in registers for the whole pass (8 chunks of K=32)
  v16bf qf[8];
  {
    const __bf16* qrow = Qp + (size_t)(m0w + ln) * DHEAD;
#pragma unroll
    for (int c = 0; c < 8; ++c) {
      v8bf lo = *reinterpret_cast<const v8bf*>(qrow + c * 32 + hi * 8);
      v8bf h8 = *reinterpret_cast<const v8bf*>(qrow + c * 32 + hi * 8 + 16);
      qf[c] = joinv(lo, h8);
    }
  }

  v8f o[16];
#pragma unroll
  for (int j = 0; j < 16; ++j) o[j] = zero8();
  float mr[8], lr[8];
#pragma unroll
  for (int r = 0; r < 8; ++r) { mr[r] = -3.0e38f; lr[r] = 0.0f; }

  // Prologue: fill buffer 0 with tile 0
  issue_k_copy(Kp, Ks[0], 0, tid);
  {
    v8bf va[2], vb[2];
    load_v_regs(Vp, 0, tid, va, vb);
    store_v_lds(Vt[0], tid, va, vb);
  }
#if HAVE_ASYNC_COPY
  WAIT_ASYNC();
#endif
  __syncthreads();

  const int NT = SEQ / 32;
  for (int kt = 0; kt < NT; ++kt) {
    const int cur = kt & 1, nxt = cur ^ 1;
    const int kb0 = kt * 32;
    const bool more = (kt + 1) < NT;

    // Kick off next tile: async K copy + V global loads (in flight over compute)
    v8bf va[2], vb[2];
    if (more) {
      issue_k_copy(Kp, Ks[nxt], kb0 + 32, tid);
      load_v_regs(Vp, kb0 + 32, tid, va, vb);
      if (kt + 2 < NT) {  // prefetch tile kt+2 (one 128B line per thread)
        const __bf16* pf = ((tid & 128) ? Vp : Kp) +
                           (size_t)(kb0 + 64) * DHEAD + (size_t)(tid & 127) * 64;
        __builtin_prefetch((const void*)pf, 0, 1);
      }
    }

    const __bf16* Kc = Ks[cur];
    const __bf16* Vc = Vt[cur];

    // S = Q . K^T : two 16x16 tiles (keys 0-15, 16-31), 8 K-chunks of 32
    v8f s0 = zero8(), s1 = zero8();
#pragma unroll
    for (int c = 0; c < 8; ++c) {
      const int fb = c * 32 + hi * 8;
      v8bf l0 = *reinterpret_cast<const v8bf*>(&Kc[ln * 264 + fb]);
      v8bf h0 = *reinterpret_cast<const v8bf*>(&Kc[ln * 264 + fb + 16]);
      s0 = __builtin_amdgcn_wmma_f32_16x16x32_bf16(
          false, qf[c], false, joinv(l0, h0), (short)0, s0, false, false);
      v8bf l1 = *reinterpret_cast<const v8bf*>(&Kc[(16 + ln) * 264 + fb]);
      v8bf h1 = *reinterpret_cast<const v8bf*>(&Kc[(16 + ln) * 264 + fb + 16]);
      s1 = __builtin_amdgcn_wmma_f32_16x16x32_bf16(
          false, qf[c], false, joinv(l1, h1), (short)0, s1, false, false);
    }

    // Online softmax (scale 1/sqrt(256) = 0.0625)
    const float sc = 0.0625f;
    float alpha[8];
#pragma unroll
    for (int r = 0; r < 8; ++r) {
      float a0 = s0[r] * sc, a1 = s1[r] * sc;
      float mx = lanemax16(fmaxf(a0, a1), lane);
      float mnew = fmaxf(mr[r], mx);
      float p0 = __expf(a0 - mnew);
      float p1 = __expf(a1 - mnew);
      alpha[r] = __expf(mr[r] - mnew);
      float rs = lanesum16(p0 + p1, lane);
      lr[r] = lr[r] * alpha[r] + rs;
      mr[r] = mnew;
      s0[r] = p0; s1[r] = p1;
    }
#pragma unroll
    for (int j = 0; j < 16; ++j) {
#pragma unroll
      for (int r = 0; r < 8; ++r) o[j][r] *= alpha[r];
    }

    // Stage P through wave-private LDS (C-fragment -> A-fragment relayout)
    __bf16* pw = &Ps[w * 16 * 40];
#pragma unroll
    for (int r = 0; r < 8; ++r) {
      int row = r + 8 * hi;
      pw[row * 40 + ln]      = (__bf16)s0[r];
      pw[row * 40 + 16 + ln] = (__bf16)s1[r];
    }
    asm volatile("s_wait_dscnt 0" ::: "memory");
    v8bf plo = *reinterpret_cast<const v8bf*>(&pw[ln * 40 + hi * 8]);
    v8bf phi = *reinterpret_cast<const v8bf*>(&pw[ln * 40 + hi * 8 + 16]);
    v16bf pfrag = joinv(plo, phi);

    // O += P . V  (16 D-chunks of 16 cols)
#pragma unroll
    for (int j = 0; j < 16; ++j) {
      const int d = j * 16 + ln;
      v8bf vlo = *reinterpret_cast<const v8bf*>(&Vc[d * 40 + hi * 8]);
      v8bf vhi = *reinterpret_cast<const v8bf*>(&Vc[d * 40 + hi * 8 + 16]);
      o[j] = __builtin_amdgcn_wmma_f32_16x16x32_bf16(
          false, pfrag, false, joinv(vlo, vhi), (short)0, o[j], false, false);
    }

    // Finish next-tile fill: V register -> LDS (other buffer), wait async K
    if (more) store_v_lds(Vt[nxt], tid, va, vb);
#if HAVE_ASYNC_COPY
    WAIT_ASYNC();
#endif
    __syncthreads();
  }

  // Epilogue: normalize and store f32 output
#pragma unroll
  for (int r = 0; r < 8; ++r) {
    float inv = 1.0f / lr[r];
    int grow = m0w + r + 8 * hi;
    float* orow = Out + ((size_t)b * SEQ + grow) * DHEAD;
#pragma unroll
    for (int j = 0; j < 16; ++j) orow[j * 16 + ln] = o[j][r] * inv;
  }
}

extern "C" void kernel_launch(void* const* d_in, const int* in_sizes, int n_in,
                              void* d_out, int out_size, void* d_ws, size_t ws_size,
                              hipStream_t stream) {
  const float* query = (const float*)d_in[0];
  const float* key   = (const float*)d_in[1];
  const float* value = (const float*)d_in[2];
  const float* WQ    = (const float*)d_in[3];
  const float* WK    = (const float*)d_in[4];
  const float* WV    = (const float*)d_in[5];
  float* out = (float*)d_out;

  const int M = BATCH * SEQ;  // 32768
  __bf16* Qb = (__bf16*)d_ws;
  __bf16* Kb = Qb + (size_t)M * DHEAD;
  __bf16* Vb = Kb + (size_t)M * DHEAD;

  dim3 pgrid(M / 128, DHEAD / 64);
  proj_kernel<<<pgrid, 256, 0, stream>>>(query, WQ, Qb);
  proj_kernel<<<pgrid, 256, 0, stream>>>(key,   WK, Kb);
  proj_kernel<<<pgrid, 256, 0, stream>>>(value, WV, Vb);

  dim3 agrid(SEQ / 128, BATCH);
  attn_kernel<<<agrid, 256, 0, stream>>>(Qb, Kb, Vb, out);
}
